// MultiHeadAttention_26998164422786
// MI455X (gfx1250) — matrix operand order repaired and checked by reading, not compiled, not run on hardware
//
#include <hip/hip_runtime.h>

// ---------------------------------------------------------------------------
// MHA forward for gfx1250 (MI455X): bf16 WMMA + TDM tensor loads, fp32 accum.
// ---------------------------------------------------------------------------

typedef __bf16 bf16;
typedef __attribute__((ext_vector_type(8)))  bf16     v8bf;
typedef __attribute__((ext_vector_type(16))) bf16     v16bf;
typedef __attribute__((ext_vector_type(8)))  float    v8f;
typedef __attribute__((ext_vector_type(4)))  unsigned u32x4;
typedef __attribute__((ext_vector_type(8)))  unsigned u32x8;

#define DIM_C   1024
#define NHEADS  16
#define HEADD   64
#define BB      4
#define TT      2048
#define MROWS   (BB * TT)        // 8192
#define L2E     1.44269504088896340f

// ---- fp32 -> bf16, round to nearest even --------------------------------
__device__ __forceinline__ bf16 f2bf(float f) {
    unsigned u = __builtin_bit_cast(unsigned, f);
    u += 0x7fffu + ((u >> 16) & 1u);
    unsigned short h = (unsigned short)(u >> 16);
    return __builtin_bit_cast(bf16, h);
}

// ---- WMMA fragment loaders (contiguous k = 16*half + s, same order for A and B, wave32) --------
__device__ __forceinline__ v16bf load_a(const bf16* p, int ld, int lane) {
    const int row = lane & 15, half = lane >> 4;
    const bf16* q = p + row * ld + half * 16;
    v8bf lo = *(const v8bf*)q;
    v8bf hi = *(const v8bf*)(q + 8);
    return __builtin_shufflevector(lo, hi, 0,1,2,3,4,5,6,7,8,9,10,11,12,13,14,15);
}

// multiplies by the stored (N x K row-major) matrix TRANSPOSED (nn.Linear/QK^T)
__device__ __forceinline__ v16bf load_b(const bf16* p, int ld, int lane) {
    const int row = lane & 15, half = lane >> 4;
    const bf16* q = p + row * ld + half * 16;
    v8bf lo = *(const v8bf*)q;
    v8bf hi = *(const v8bf*)(q + 8);
    return __builtin_shufflevector(lo, hi, 0,1,2,3,4,5,6,7,8,9,10,11,12,13,14,15);
}

// B operand from a K x N row-major matrix (V in P.V): column gather from LDS
__device__ __forceinline__ v16bf load_bt(const bf16* p, int ld, int lane) {
    const int n = lane & 15, half = lane >> 4;
    v16bf r;
#pragma unroll
    for (int e = 0; e < 16; ++e) r[e] = p[(half * 16 + e) * ld + n];
    return r;
}

__device__ __forceinline__ v8f wmma_bf16(v16bf a, v16bf b, v8f c) {
    return __builtin_amdgcn_wmma_f32_16x16x32_bf16(
        false, a, false, b, (short)0, c, false, false);
}

template <int N>
__device__ __forceinline__ void wait_tensorcnt() {
#if __has_builtin(__builtin_amdgcn_s_wait_tensorcnt)
    __builtin_amdgcn_s_wait_tensorcnt(N);
#else
    asm volatile("s_wait_tensorcnt %0" :: "i"(N) : "memory");
#endif
}

// ---- TDM: DMA one 32(rows) x 64(bf16) tile, row stride 1024 elems, into LDS
// with padding 8 bf16 per row (LDS row stride 72). D# per CDNA5 ISA sec. 8.
__device__ __forceinline__ void tdm_load_tile(unsigned lds_addr, const bf16* gp) {
    unsigned long long ga = (unsigned long long)(__UINTPTR_TYPE__)gp;
    u32x4 g0 = { 1u,                                   // count=1, user mode
                 lds_addr,                             // LDS byte address
                 (unsigned)ga,                         // global_addr[31:0]
                 (unsigned)((ga >> 32) & 0x1FFFFFFu)   // global_addr[56:32]
                   | (2u << 30) };                     // type=2 (image)
    u32x8 g1 = { (1u << 16)      // data_size = 1 (2 bytes)
                   | (1u << 20)  // pad_enable
                   | (4u << 22)  // pad_interval: 32 DWORDs (= one 128B row)
                   | (3u << 25), // pad_amount: 4 DWORDs (= 8 bf16)  -> LK=72
                 (64u << 16),    // tensor_dim0 = 64 -> bits[79:48]
                 (32u << 16),    // tensor_dim1 = 32 -> bits[111:80]
                 (64u << 16),    // tile_dim0 = 64   -> bits[127:112]
                 32u,            // tile_dim1 = 32, tile_dim2 = 0
                 1024u,          // tensor_dim0_stride = 1024 elems
                 0u, 0u };
    u32x4 gz = { 0u, 0u, 0u, 0u };
    asm volatile("tensor_load_to_lds %0, %1, %2, %3"
                 :: "s"(g0), "s"(g1), "s"(gz), "s"(gz)
                 : "memory");
}

// ---- elementwise fp32 -> bf16 convert ------------------------------------
__global__ void cvt_bf16_kernel(const float* __restrict__ in,
                                bf16* __restrict__ out, int n) {
    for (int i = blockIdx.x * blockDim.x + threadIdx.x; i < n;
         i += gridDim.x * blockDim.x)
        out[i] = f2bf(in[i]);
}

// ---- software-pipelined GEMM core: wave tile 32(M) x 64(N) ---------------
// Two-stage ping-pong in distinct registers (no rotation copies): K step 64,
// stage X covers k0, stage Y covers k0+32; X reloads k0+64 while Y computes.
struct GemmAcc { v8f a[2][4]; };

__device__ __forceinline__ GemmAcc gemm_tile_32x64(const bf16* __restrict__ A,
                                                   const bf16* __restrict__ W,
                                                   int m0, int n0, int lane) {
    GemmAcc acc;
#pragma unroll
    for (int i = 0; i < 2; ++i)
#pragma unroll
        for (int t = 0; t < 4; ++t) acc.a[i][t] = {};

    const bf16* Ar0 = A + (long)m0 * DIM_C;
    const bf16* Ar1 = A + (long)(m0 + 16) * DIM_C;
    const bf16* W0  = W + (long)(n0 +  0) * DIM_C;
    const bf16* W1  = W + (long)(n0 + 16) * DIM_C;
    const bf16* W2  = W + (long)(n0 + 32) * DIM_C;
    const bf16* W3  = W + (long)(n0 + 48) * DIM_C;

    // stage X: k = 0
    v16bf xA0 = load_a(Ar0, DIM_C, lane), xA1 = load_a(Ar1, DIM_C, lane);
    v16bf xB0 = load_b(W0, DIM_C, lane),  xB1 = load_b(W1, DIM_C, lane);
    v16bf xB2 = load_b(W2, DIM_C, lane),  xB3 = load_b(W3, DIM_C, lane);

#pragma unroll 1
    for (int k0 = 0; k0 < DIM_C; k0 += 64) {
        const int ky = k0 + 32;                    // stage Y k-index
        const int kx = (k0 + 64) & (DIM_C - 1);    // next X (wraps, branch-free)

        // load stage Y while X computes
        v16bf yA0 = load_a(Ar0 + ky, DIM_C, lane), yA1 = load_a(Ar1 + ky, DIM_C, lane);
        v16bf yB0 = load_b(W0 + ky, DIM_C, lane),  yB1 = load_b(W1 + ky, DIM_C, lane);
        v16bf yB2 = load_b(W2 + ky, DIM_C, lane),  yB3 = load_b(W3 + ky, DIM_C, lane);

        acc.a[0][0] = wmma_bf16(xA0, xB0, acc.a[0][0]);
        acc.a[1][0] = wmma_bf16(xA1, xB0, acc.a[1][0]);
        acc.a[0][1] = wmma_bf16(xA0, xB1, acc.a[0][1]);
        acc.a[1][1] = wmma_bf16(xA1, xB1, acc.a[1][1]);
        acc.a[0][2] = wmma_bf16(xA0, xB2, acc.a[0][2]);
        acc.a[1][2] = wmma_bf16(xA1, xB2, acc.a[1][2]);
        acc.a[0][3] = wmma_bf16(xA0, xB3, acc.a[0][3]);
        acc.a[1][3] = wmma_bf16(xA1, xB3, acc.a[1][3]);

        // reload stage X (k0+64) while Y computes (harmless wrap on last step)
        xA0 = load_a(Ar0 + kx, DIM_C, lane); xA1 = load_a(Ar1 + kx, DIM_C, lane);
        xB0 = load_b(W0 + kx, DIM_C, lane);  xB1 = load_b(W1 + kx, DIM_C, lane);
        xB2 = load_b(W2 + kx, DIM_C, lane);  xB3 = load_b(W3 + kx, DIM_C, lane);

        acc.a[0][0] = wmma_bf16(yA0, yB0, acc.a[0][0]);
        acc.a[1][0] = wmma_bf16(yA1, yB0, acc.a[1][0]);
        acc.a[0][1] = wmma_bf16(yA0, yB1, acc.a[0][1]);
        acc.a[1][1] = wmma_bf16(yA1, yB1, acc.a[1][1]);
        acc.a[0][2] = wmma_bf16(yA0, yB2, acc.a[0][2]);
        acc.a[1][2] = wmma_bf16(yA1, yB2, acc.a[1][2]);
        acc.a[0][3] = wmma_bf16(yA0, yB3, acc.a[0][3]);
        acc.a[1][3] = wmma_bf16(yA1, yB3, acc.a[1][3]);
    }
    return acc;
}

// ---- fused QKV projection -------------------------------------------------
// grid (M/128, N/64, 3), block 128 (4 waves). Wave tile: 32(M) x 64(N).
__global__ void __launch_bounds__(128)
qkv_gemm_kernel(const bf16* __restrict__ X,
                const bf16* __restrict__ Wq, const bf16* __restrict__ Wk,
                const bf16* __restrict__ Wv,
                bf16* __restrict__ Qo, bf16* __restrict__ Ko,
                bf16* __restrict__ Vo) {
    const int lane = threadIdx.x & 31, wave = threadIdx.x >> 5;
    const int m0 = blockIdx.x * 128 + wave * 32;
    const int n0 = blockIdx.y * 64;

    const bf16* W; bf16* Out; float scale;
    if (blockIdx.z == 0)      { W = Wq; Out = Qo; scale = 0.125f; } // 1/sqrt(64)
    else if (blockIdx.z == 1) { W = Wk; Out = Ko; scale = 1.0f; }
    else                      { W = Wv; Out = Vo; scale = 1.0f; }

    GemmAcc acc = gemm_tile_32x64(X, W, m0, n0, lane);

    const int col = n0 + (lane & 15);
    const int rsub = (lane & 16) ? 8 : 0;
#pragma unroll
    for (int i = 0; i < 2; ++i)
#pragma unroll
        for (int e = 0; e < 8; ++e) {
            const long r = m0 + i * 16 + rsub + e;
#pragma unroll
            for (int t = 0; t < 4; ++t)
                Out[r * DIM_C + col + t * 16] = f2bf(acc.a[i][t][e] * scale);
        }
}

// ---- flash attention (causal) --------------------------------------------
// grid (T/64, B*H), block 128 (4 waves x 16 queries). K/V tiles DMA'd by the
// TDM into double-buffered LDS; issue tile i+1, wait TENSORcnt<=2, compute i.
#define LK 72   // LDS row stride (bf16): 128B row + 16B TDM pad
#define LP 40   // per-wave P tile row stride

__global__ void __launch_bounds__(128)
attn_kernel(const bf16* __restrict__ Q, const bf16* __restrict__ K,
            const bf16* __restrict__ V, bf16* __restrict__ O) {
    __shared__ __align__(16) bf16 sK[2][32 * LK];
    __shared__ __align__(16) bf16 sV[2][32 * LK];
    __shared__ __align__(16) bf16 sP[4][16 * LP];

    const int lane = threadIdx.x & 31, wave = threadIdx.x >> 5;
    const int b = blockIdx.y >> 4, h = blockIdx.y & 15;
    const int qb0 = blockIdx.x * 64;
    const int q0  = qb0 + wave * 16;
    const int col = lane & 15;
    const int rb  = (lane & 16) ? 8 : 0;

    const bf16* Kbase = K + ((long)b * TT) * DIM_C + h * HEADD;
    const bf16* Vbase = V + ((long)b * TT) * DIM_C + h * HEADD;

    const bf16* Qp = Q + ((long)(b * TT + q0)) * DIM_C + h * HEADD;
    const v16bf qa0 = load_a(Qp,      DIM_C, lane);   // d = 0..31
    const v16bf qa1 = load_a(Qp + 32, DIM_C, lane);   // d = 32..63

    float m[8], l[8];
    v8f o0 = {}, o1 = {}, o2 = {}, o3 = {};
#pragma unroll
    for (int e = 0; e < 8; ++e) { m[e] = -__builtin_inff(); l[e] = 0.0f; }

    const int nIter = qb0 / 32 + 2;            // key tiles of 32, up to qb0+63

    if (wave == 0) {                           // prologue: DMA tile 0
        tdm_load_tile((unsigned)(__UINTPTR_TYPE__)&sK[0][0], Kbase);
        tdm_load_tile((unsigned)(__UINTPTR_TYPE__)&sV[0][0], Vbase);
    }

    for (int it = 0; it < nIter; ++it) {
        const int j0  = it * 32;
        const int buf = it & 1;
        if (wave == 0) {
            if (it + 1 < nIter) {              // DMA next tile into other buf
                tdm_load_tile((unsigned)(__UINTPTR_TYPE__)&sK[buf ^ 1][0],
                              Kbase + (long)(j0 + 32) * DIM_C);
                tdm_load_tile((unsigned)(__UINTPTR_TYPE__)&sV[buf ^ 1][0],
                              Vbase + (long)(j0 + 32) * DIM_C);
                wait_tensorcnt<2>();           // TDM in-order: tile `it` done
            } else {
                wait_tensorcnt<0>();
            }
        }
        __syncthreads();                       // publish tile `it` to all waves

        if (j0 <= q0 + 15) {
            const bf16* tK = &sK[buf][0];
            const bf16* tV = &sV[buf][0];

            // S = Q . K^T  (two 16x16 tiles, K-dim 64 = 2 WMMAs each)
            v8f s0 = {}, s1 = {};
            s0 = wmma_bf16(qa0, load_b(tK,                LK, lane), s0);
            s0 = wmma_bf16(qa1, load_b(tK + 32,           LK, lane), s0);
            s1 = wmma_bf16(qa0, load_b(tK + 16 * LK,      LK, lane), s1);
            s1 = wmma_bf16(qa1, load_b(tK + 16 * LK + 32, LK, lane), s1);

            // causal mask + row max (row lives across a 16-lane half)
            float mn[8];
#pragma unroll
            for (int e = 0; e < 8; ++e) {
                const int qg = q0 + rb + e;
                const float a0 = (j0 + col      <= qg) ? s0[e] : -__builtin_inff();
                const float a1 = (j0 + 16 + col <= qg) ? s1[e] : -__builtin_inff();
                s0[e] = a0; s1[e] = a1;
                mn[e] = fmaxf(m[e], fmaxf(a0, a1));
            }
#pragma unroll
            for (int off = 1; off <= 8; off <<= 1)
#pragma unroll
                for (int e = 0; e < 8; ++e)
                    mn[e] = fmaxf(mn[e], __shfl_xor(mn[e], off, 32));

            // exp, rescale, stage P (bf16) in per-wave LDS strip
            float rs[8];
#pragma unroll
            for (int e = 0; e < 8; ++e) {
                const float al = exp2f((m[e] - mn[e]) * L2E);
                const float p0 = exp2f((s0[e] - mn[e]) * L2E);
                const float p1 = exp2f((s1[e] - mn[e]) * L2E);
                o0[e] *= al; o1[e] *= al; o2[e] *= al; o3[e] *= al;
                l[e] *= al;
                rs[e] = p0 + p1;
                m[e]  = mn[e];
                sP[wave][(rb + e) * LP + col]      = f2bf(p0);
                sP[wave][(rb + e) * LP + 16 + col] = f2bf(p1);
            }
#pragma unroll
            for (int off = 1; off <= 8; off <<= 1)
#pragma unroll
                for (int e = 0; e < 8; ++e)
                    rs[e] += __shfl_xor(rs[e], off, 32);
#pragma unroll
            for (int e = 0; e < 8; ++e) l[e] += rs[e];

            // O += P . V
            const v16bf pa = load_a(sP[wave], LP, lane);
            o0 = wmma_bf16(pa, load_bt(tV,      LK, lane), o0);
            o1 = wmma_bf16(pa, load_bt(tV + 16, LK, lane), o1);
            o2 = wmma_bf16(pa, load_bt(tV + 32, LK, lane), o2);
            o3 = wmma_bf16(pa, load_bt(tV + 48, LK, lane), o3);
        }
        __syncthreads();    // all waves done with `buf` before TDM reuses it
    }

    bf16* Op = O + ((long)(b * TT + q0)) * DIM_C + h * HEADD;
#pragma unroll
    for (int e = 0; e < 8; ++e) {
        const float inv = 1.0f / l[e];
        const long r = rb + e;
        Op[r * DIM_C + col]      = f2bf(o0[e] * inv);
        Op[r * DIM_C + col + 16] = f2bf(o1[e] * inv);
        Op[r * DIM_C + col + 32] = f2bf(o2[e] * inv);
        Op[r * DIM_C + col + 48] = f2bf(o3[e] * inv);
    }
}

// ---- output projection: out = attn @ Wo^T + bo  (fp32 out) ---------------
__global__ void __launch_bounds__(128)
proj_gemm_kernel(const bf16* __restrict__ A, const bf16* __restrict__ Wo,
                 const float* __restrict__ bias, float* __restrict__ out) {
    const int lane = threadIdx.x & 31, wave = threadIdx.x >> 5;
    const int m0 = blockIdx.x * 128 + wave * 32;
    const int n0 = blockIdx.y * 64;

    GemmAcc acc = gemm_tile_32x64(A, Wo, m0, n0, lane);

    const int col = n0 + (lane & 15);
    const int rsub = (lane & 16) ? 8 : 0;
#pragma unroll
    for (int i = 0; i < 2; ++i)
#pragma unroll
        for (int e = 0; e < 8; ++e) {
            const long r = m0 + i * 16 + rsub + e;
#pragma unroll
            for (int t = 0; t < 4; ++t)
                out[r * DIM_C + col + t * 16] = acc.a[i][t][e] + bias[col + t * 16];
        }
}

// ---------------------------------------------------------------------------
extern "C" void kernel_launch(void* const* d_in, const int* in_sizes, int n_in,
                              void* d_out, int out_size, void* d_ws, size_t ws_size,
                              hipStream_t stream) {
    (void)in_sizes; (void)n_in; (void)out_size; (void)ws_size;
    // inputs: x, mask(unused: causal), Wq, Wk, Wv, Wo, bo, start_pos(=0)
    const float* x  = (const float*)d_in[0];
    const float* Wq = (const float*)d_in[2];
    const float* Wk = (const float*)d_in[3];
    const float* Wv = (const float*)d_in[4];
    const float* Wo = (const float*)d_in[5];
    const float* bo = (const float*)d_in[6];
    float* out = (float*)d_out;

    char* ws = (char*)d_ws;
    const size_t szX = (size_t)MROWS * DIM_C * sizeof(bf16);   // 16 MB
    const size_t szW = (size_t)DIM_C * DIM_C * sizeof(bf16);   //  2 MB
    bf16* xbf   = (bf16*)(ws);
    bf16* wqbf  = (bf16*)(ws + szX);
    bf16* wkbf  = (bf16*)(ws + szX + szW);
    bf16* wvbf  = (bf16*)(ws + szX + 2 * szW);
    bf16* wobf  = (bf16*)(ws + szX + 3 * szW);
    bf16* qbf   = (bf16*)(ws + szX + 4 * szW);
    bf16* kbf   = (bf16*)(ws + 2 * szX + 4 * szW);
    bf16* vbf   = (bf16*)(ws + 3 * szX + 4 * szW);
    bf16* attnb = (bf16*)(ws + 4 * szX + 4 * szW);             // total ~88 MB

    const int nX = MROWS * DIM_C, nW = DIM_C * DIM_C;
    cvt_bf16_kernel<<<4096, 256, 0, stream>>>(x,  xbf,  nX);
    cvt_bf16_kernel<<<2048, 256, 0, stream>>>(Wq, wqbf, nW);
    cvt_bf16_kernel<<<2048, 256, 0, stream>>>(Wk, wkbf, nW);
    cvt_bf16_kernel<<<2048, 256, 0, stream>>>(Wv, wvbf, nW);
    cvt_bf16_kernel<<<2048, 256, 0, stream>>>(Wo, wobf, nW);

    qkv_gemm_kernel<<<dim3(MROWS / 128, DIM_C / 64, 3), 128, 0, stream>>>(
        xbf, wqbf, wkbf, wvbf, qbf, kbf, vbf);

    attn_kernel<<<dim3(TT / 64, BB * NHEADS), 128, 0, stream>>>(
        qbf, kbf, vbf, attnb);

    proj_gemm_kernel<<<dim3(MROWS / 128, DIM_C / 64, 1), 128, 0, stream>>>(
        attnb, wobf, bo, out);
}
